// Decoder_79413945303935
// MI455X (gfx1250) — matrix operand order repaired and checked by reading, not compiled, not run on hardware
//
#include <hip/hip_runtime.h>

typedef __attribute__((ext_vector_type(16))) __bf16 v16bf;
typedef __attribute__((ext_vector_type(8)))  float  v8f;

#define T_   2000
#define N_   128
#define L_   250
#define H_   512
#define K_   128
#define V_   128
#define A_   64
#define TPAD 2048
#define K1   1152   // [emb(512) | ctx(128) | h1(512)]  -> LSTM1 A columns
#define G1   2048   // 4*H
#define K2   640    // [h1(512) | h2(128)]              -> LSTM2 A columns
#define G2   512    // 4*K
#define KM   256    // [h2(128) | ctx(128)]             -> MoS A columns
#define NSLICE 4
#define TS   (T_ / NSLICE)          // 500 attention timesteps per slice WG
#define PARTW 2176                  // floats per (tile,slice) partial record

// ---------------------------------------------------------------------------
// helpers
// ---------------------------------------------------------------------------
__device__ __forceinline__ float bf2f(unsigned us) {
    union { unsigned u; float f; } x; x.u = us << 16; return x.f;
}
__device__ __forceinline__ float sigm(float x) { return 1.f / (1.f + __expf(-x)); }

// Load a 16x32 bf16 A-fragment (WMMA layout, ISA 7.12.2) from row-major LDS.
// lane l: M = l&15 ; VGPR v holds K pair  (v>=4?16:0)+(l>=16?8:0)+(v&3)*2
__device__ __forceinline__ v16bf load_a_frag(const __bf16* base, int strideK, int lane) {
    const int m = lane & 15;
    const int khalf = (lane >= 16) ? 8 : 0;
    const unsigned* p = reinterpret_cast<const unsigned*>(base + m * strideK);
    union { v16bf v; unsigned u[8]; } r;
#pragma unroll
    for (int v = 0; v < 8; ++v) {
        const int k = ((v & 4) ? 16 : 0) + khalf + (v & 3) * 2;
        r.u[v] = p[k >> 1];
    }
    return r.v;
}

// ---------------------------------------------------------------------------
// one-time prep kernels
// ---------------------------------------------------------------------------
__global__ void zero_bar(int* __restrict__ bar, int n) {
    const int i = blockIdx.x * blockDim.x + threadIdx.x;
    if (i < n) bar[i] = 0;
}

// Pack W = [A | B] (rows = out dim n, cols = k) into per-lane WMMA B-fragment
// order: frag(nt,kt) holds 512 bf16; element = lane*16 + v*2 + p with
// k_local = (v>=4?16:0) + (lane>=16?8:0) + 2(v&3) + p , n_local = lane&15.
__global__ void pack_wpair(const float* __restrict__ A, int ka,
                           const float* __restrict__ B, int kb,
                           __bf16* __restrict__ outp, int ntiles, int ktiles) {
    const int total = ntiles * ktiles * 512;
    const int idx = blockIdx.x * blockDim.x + threadIdx.x;
    if (idx >= total) return;
    const int e = idx & 511, f = idx >> 9;
    const int kt = f % ktiles, nt = f / ktiles;
    const int lane = e >> 4, rem = e & 15, v = rem >> 1, p = rem & 1;
    const int kl = ((v >= 4) ? 16 : 0) + ((lane >= 16) ? 8 : 0) + 2 * (v & 3) + p;
    const int n = nt * 16 + (lane & 15);
    const int k = kt * 32 + kl;
    const float val = (k < ka) ? A[(size_t)n * ka + k] : B[(size_t)n * kb + (k - ka)];
    outp[idx] = (__bf16)val;
}

// key (T,N,K) f32 -> keyT2[n][k2][t] : packed pair (k=2k2, 2k2+1) as bf16x2
__global__ void tkey(const float* __restrict__ key, unsigned* __restrict__ keyT2) {
    const int idx = blockIdx.x * blockDim.x + threadIdx.x;
    if (idx >= N_ * (K_ / 2) * TPAD) return;
    const int t = idx & (TPAD - 1);
    const int k2 = (idx >> 11) & 63;
    const int n = idx >> 17;
    unsigned r = 0;
    if (t < T_) {
        const size_t base = ((size_t)t * N_ + n) * K_ + 2 * k2;
        const __bf16 lo = (__bf16)key[base];
        const __bf16 hi = (__bf16)key[base + 1];
        unsigned short ul, uh;
        __builtin_memcpy(&ul, &lo, 2);
        __builtin_memcpy(&uh, &hi, 2);
        r = (unsigned)ul | ((unsigned)uh << 16);
    }
    keyT2[idx] = r;
}

// values (T,N,V) f32 -> valT[n][t][v] bf16
__global__ void tval(const float* __restrict__ values, __bf16* __restrict__ valT) {
    const int idx = blockIdx.x * blockDim.x + threadIdx.x;
    if (idx >= N_ * TPAD * V_) return;
    const int v = idx & 127;
    const int t = (idx >> 7) & (TPAD - 1);
    const int n = idx >> 18;
    float val = 0.f;
    if (t < T_) val = values[((size_t)t * N_ + n) * V_ + v];
    valT[idx] = (__bf16)val;
}

// ---------------------------------------------------------------------------
// persistent decoder: 8 batch tiles x 4 attention T-slices = 32 workgroups.
// Every slice WG redundantly runs the LSTM stack (streams weights from L2,
// no state exchange); attention is T-split with a flash-style combine through
// global memory, one spin barrier per step.
// ---------------------------------------------------------------------------
__global__ __launch_bounds__(512, 1)
void decoder_persistent(const int* __restrict__ lens, const int* __restrict__ text,
                        const float* __restrict__ emb, const float* __restrict__ values0,
                        const float* __restrict__ b_ih1, const float* __restrict__ b_hh1,
                        const float* __restrict__ b_ih2, const float* __restrict__ b_hh2,
                        const float* __restrict__ b_mos,
                        const __bf16* __restrict__ w1p, const __bf16* __restrict__ w2p,
                        const __bf16* __restrict__ wmp,
                        const unsigned* __restrict__ keyT2, const __bf16* __restrict__ valT,
                        float* __restrict__ part, int* __restrict__ bar,
                        float* __restrict__ out) {
    __shared__ float  big[16 * G1];       // gates1 / gates2 / slice scores (reused)
    __shared__ __bf16 aS[16 * K1];        // LSTM1 A rows [emb|ctx|h1]
    __shared__ __bf16 a2S[16 * K2];       // LSTM2 A rows [h1|h2]
    __shared__ __bf16 amS[16 * KM];       // MoS   A rows [h2|ctx]
    __shared__ float  c1S[16 * H_];
    __shared__ float  c2S[16 * K_];
    __shared__ float  ctxS[16 * V_];
    __shared__ float  h2fS[16 * K_];
    __shared__ float  facS[16 * NSLICE];  // combine factors e^{m_i-M}/S
    __shared__ int    lensS[16];

    const int tid  = threadIdx.x;
    const int wave = tid >> 5;
    const int lane = tid & 31;
    const int tb   = blockIdx.x >> 2;     // batch tile 0..7
    const int sl   = blockIdx.x & 3;      // attention T-slice 0..3
    const int n0   = tb * 16;
    const int t0   = sl * TS;

    // ---- init persistent state --------------------------------------------
    for (int idx = tid; idx < 16 * H_; idx += 512) {
        c1S[idx] = 0.f;
        const int n = idx >> 9, j = idx & 511;
        aS[n * K1 + 640 + j] = (__bf16)0.f;              // h1 = 0
    }
    for (int idx = tid; idx < 16 * K_; idx += 512) {
        const int n = idx >> 7, j = idx & 127;
        c2S[idx] = 0.f;
        a2S[n * K2 + 512 + j] = (__bf16)0.f;             // h2 = 0
        ctxS[idx] = values0[(size_t)(n0 + n) * V_ + j];  // ctx = values[0]
    }
    if (tid < 16) lensS[tid] = lens[n0 + tid];
    __syncthreads();

    for (int l = 0; l < L_; ++l) {
        // ---- phase 1: stage [emb | ctx] into A1 ---------------------------
        for (int n = 0; n < 16; ++n) {
            const int tok = text[(size_t)(n0 + n) * L_ + l];
            aS[n * K1 + tid] = (tok != 0) ? (__bf16)emb[(size_t)tok * H_ + tid] : (__bf16)0.f;
        }
        for (int idx = tid; idx < 16 * V_; idx += 512) {
            const int n = idx >> 7, v = idx & 127;
            aS[n * K1 + 512 + v] = (__bf16)ctxS[idx];
        }
        __syncthreads();

        // ---- phase 2: LSTM1 GEMM  (16 x 2048, K=1152) ---------------------
        {
            v8f acc[8];
#pragma unroll
            for (int i = 0; i < 8; ++i) acc[i] = (v8f){0.f, 0.f, 0.f, 0.f, 0.f, 0.f, 0.f, 0.f};
            for (int kt = 0; kt < K1 / 32; ++kt) {
                const v16bf a = load_a_frag(aS + kt * 32, K1, lane);
                v16bf bfr[8];
#pragma unroll
                for (int i = 0; i < 8; ++i)
                    bfr[i] = *(const v16bf*)(w1p +
                        ((size_t)((wave * 8 + i) * (K1 / 32) + kt)) * 512 + lane * 16);
#pragma unroll
                for (int i = 0; i < 8; ++i)
                    acc[i] = __builtin_amdgcn_wmma_f32_16x16x32_bf16(
                        false, a, false, bfr[i], (short)0, acc[i], false, false);
            }
            const int col = lane & 15, mb = (lane >> 4) << 3;
#pragma unroll
            for (int i = 0; i < 8; ++i) {
                const int nc = (wave * 8 + i) * 16 + col;
#pragma unroll
                for (int r = 0; r < 8; ++r) big[(r + mb) * G1 + nc] = acc[i][r];
            }
        }
        __syncthreads();

        // ---- phase 3: LSTM1 elementwise -----------------------------------
        for (int idx = tid; idx < 16 * H_; idx += 512) {
            const int n = idx >> 9, j = idx & 511;
            const float gi = big[n * G1 + j]        + b_ih1[j]        + b_hh1[j];
            const float gf = big[n * G1 + 512 + j]  + b_ih1[512 + j]  + b_hh1[512 + j];
            const float gg = big[n * G1 + 1024 + j] + b_ih1[1024 + j] + b_hh1[1024 + j];
            const float go = big[n * G1 + 1536 + j] + b_ih1[1536 + j] + b_hh1[1536 + j];
            const float c = sigm(gf) * c1S[idx] + sigm(gi) * tanhf(gg);
            const float h = sigm(go) * tanhf(c);
            c1S[idx] = c;
            aS[n * K1 + 640 + j] = (__bf16)h;
            a2S[n * K2 + j]      = (__bf16)h;
        }
        __syncthreads();

        // ---- phase 4: LSTM2 GEMM (16 x 512, K=640) ------------------------
        {
            v8f acc[2];
#pragma unroll
            for (int i = 0; i < 2; ++i) acc[i] = (v8f){0.f, 0.f, 0.f, 0.f, 0.f, 0.f, 0.f, 0.f};
            for (int kt = 0; kt < K2 / 32; ++kt) {
                const v16bf a = load_a_frag(a2S + kt * 32, K2, lane);
                v16bf bfr[2];
#pragma unroll
                for (int i = 0; i < 2; ++i)
                    bfr[i] = *(const v16bf*)(w2p +
                        ((size_t)((wave * 2 + i) * (K2 / 32) + kt)) * 512 + lane * 16);
#pragma unroll
                for (int i = 0; i < 2; ++i)
                    acc[i] = __builtin_amdgcn_wmma_f32_16x16x32_bf16(
                        false, a, false, bfr[i], (short)0, acc[i], false, false);
            }
            const int col = lane & 15, mb = (lane >> 4) << 3;
#pragma unroll
            for (int i = 0; i < 2; ++i) {
                const int nc = (wave * 2 + i) * 16 + col;
#pragma unroll
                for (int r = 0; r < 8; ++r) big[(r + mb) * G2 + nc] = acc[i][r];
            }
        }
        __syncthreads();

        // ---- phase 5: LSTM2 elementwise -----------------------------------
        for (int idx = tid; idx < 16 * K_; idx += 512) {
            const int n = idx >> 7, j = idx & 127;
            const float gi = big[n * G2 + j]        + b_ih2[j]        + b_hh2[j];
            const float gf = big[n * G2 + 128 + j]  + b_ih2[128 + j]  + b_hh2[128 + j];
            const float gg = big[n * G2 + 256 + j]  + b_ih2[256 + j]  + b_hh2[256 + j];
            const float go = big[n * G2 + 384 + j]  + b_ih2[384 + j]  + b_hh2[384 + j];
            const float c = sigm(gf) * c2S[idx] + sigm(gi) * tanhf(gg);
            const float h = sigm(go) * tanhf(c);
            c2S[idx] = c;
            h2fS[idx] = h;
            a2S[n * K2 + 512 + j] = (__bf16)h;
            amS[n * KM + j]       = (__bf16)h;
        }
        __syncthreads();

        // ---- phase 6: slice attention scores (bf16 key stream from L2) ----
        for (int tt = tid; tt < TS; tt += 512) {
            const int t = t0 + tt;
            for (int n = 0; n < 16; ++n) {
                const unsigned* kp = keyT2 + ((size_t)(n0 + n) * 64) * TPAD + t;
                float s = 0.f;
#pragma unroll 8
                for (int k2 = 0; k2 < 64; ++k2) {
                    const unsigned u = kp[(size_t)k2 * TPAD];
                    s += bf2f(u & 0xffffu) * h2fS[n * K_ + 2 * k2]
                       + bf2f(u >> 16)     * h2fS[n * K_ + 2 * k2 + 1];
                }
                big[n * 512 + tt] = (t >= lensS[n]) ? -1e9f : s;
            }
        }
        __syncthreads();

        // ---- phase 7+8: slice-local softmax partial + unnormalized ctx ----
        {
            const int n = wave;                      // one wave per batch row
            float m = -3.4e38f;
            for (int tt = lane; tt < TS; tt += 32) m = fmaxf(m, big[n * 512 + tt]);
#pragma unroll
            for (int off = 16; off > 0; off >>= 1) m = fmaxf(m, __shfl_xor(m, off, 32));
            float s = 0.f;
            for (int tt = lane; tt < TS; tt += 32) {
                const float e = __expf(big[n * 512 + tt] - m);
                big[n * 512 + tt] = e;
                s += e;
            }
#pragma unroll
            for (int off = 16; off > 0; off >>= 1) s += __shfl_xor(s, off, 32);

            const __bf16* vp = valT + ((size_t)(n0 + n) * TPAD + t0) * V_ + lane * 4;
            float a0 = 0.f, a1 = 0.f, a2 = 0.f, a3 = 0.f;
            for (int tt = 0; tt < TS; ++tt) {
                const float w = big[n * 512 + tt];
                const uint2 q = *(const uint2*)(vp + (size_t)tt * V_);
                a0 += w * bf2f(q.x & 0xffffu);
                a1 += w * bf2f(q.x >> 16);
                a2 += w * bf2f(q.y & 0xffffu);
                a3 += w * bf2f(q.y >> 16);
            }
            float* pp = part + (size_t)(tb * NSLICE + sl) * PARTW;
            if (lane == 0) { pp[n] = m; pp[16 + n] = s; }
            const int v = lane * 4;
            pp[32 + n * V_ + v + 0] = a0;
            pp[32 + n * V_ + v + 1] = a1;
            pp[32 + n * V_ + v + 2] = a2;
            pp[32 + n * V_ + v + 3] = a3;
        }
        // ---- publish + one spin barrier per step --------------------------
        __threadfence();
        __syncthreads();
        if (tid == 0) {
            int* ctr = bar + tb * L_ + l;
            __hip_atomic_fetch_add(ctr, 1, __ATOMIC_RELEASE, __HIP_MEMORY_SCOPE_AGENT);
            while (__hip_atomic_load(ctr, __ATOMIC_ACQUIRE, __HIP_MEMORY_SCOPE_AGENT) < NSLICE)
                __builtin_amdgcn_s_sleep(8);
        }
        __syncthreads();
        __threadfence();

        // ---- combine partials (done redundantly by all 4 slice WGs) -------
        if (tid < 16) {
            const int n = tid;
            float mi[NSLICE], si[NSLICE];
#pragma unroll
            for (int i = 0; i < NSLICE; ++i) {
                const float* q = part + (size_t)(tb * NSLICE + i) * PARTW;
                mi[i] = q[n]; si[i] = q[16 + n];
            }
            float M = mi[0];
#pragma unroll
            for (int i = 1; i < NSLICE; ++i) M = fmaxf(M, mi[i]);
            float S = 0.f;
            float f[NSLICE];
#pragma unroll
            for (int i = 0; i < NSLICE; ++i) { f[i] = __expf(mi[i] - M); S += si[i] * f[i]; }
            const float invS = 1.f / S;
#pragma unroll
            for (int i = 0; i < NSLICE; ++i) facS[n * NSLICE + i] = f[i] * invS;
        }
        __syncthreads();
        for (int idx = tid; idx < 16 * V_; idx += 512) {
            const int n = idx >> 7, v = idx & 127;
            float c = 0.f;
#pragma unroll
            for (int i = 0; i < NSLICE; ++i)
                c += part[(size_t)(tb * NSLICE + i) * PARTW + 32 + idx] * facS[n * NSLICE + i];
            ctxS[idx] = c;
            amS[n * KM + 128 + v] = (__bf16)c;
        }
        __syncthreads();

        // ---- phase 9: MoS projection (slice 0 only writes output) ---------
        if (sl == 0 && wave < 4) {
            v8f acc = (v8f){0.f, 0.f, 0.f, 0.f, 0.f, 0.f, 0.f, 0.f};
            for (int kt = 0; kt < KM / 32; ++kt) {
                const v16bf a = load_a_frag(amS + kt * 32, KM, lane);
                const v16bf b = *(const v16bf*)(wmp +
                    ((size_t)(wave * (KM / 32) + kt)) * 512 + lane * 16);
                acc = __builtin_amdgcn_wmma_f32_16x16x32_bf16(
                    false, a, false, b, (short)0, acc, false, false);
            }
            const int colA = wave * 16 + (lane & 15);
            const float bb = b_mos[colA];
            const int mb = (lane >> 4) << 3;
#pragma unroll
            for (int r = 0; r < 8; ++r) {
                const int m = r + mb;
                out[((size_t)(n0 + m) * L_ + l) * A_ + colA] = acc[r] + bb;
            }
        }
        __syncthreads();
    }
}

// ---------------------------------------------------------------------------
extern "C" void kernel_launch(void* const* d_in, const int* in_sizes, int n_in,
                              void* d_out, int out_size, void* d_ws, size_t ws_size,
                              hipStream_t stream) {
    (void)in_sizes; (void)n_in; (void)out_size; (void)ws_size;
    const float* key    = (const float*)d_in[0];
    const float* values = (const float*)d_in[1];
    const int*   lens   = (const int*)d_in[2];
    const int*   text   = (const int*)d_in[3];
    const float* emb    = (const float*)d_in[4];
    const float* W_ih1  = (const float*)d_in[5];
    const float* W_hh1  = (const float*)d_in[6];
    const float* b_ih1  = (const float*)d_in[7];
    const float* b_hh1  = (const float*)d_in[8];
    const float* W_ih2  = (const float*)d_in[9];
    const float* W_hh2  = (const float*)d_in[10];
    const float* b_ih2  = (const float*)d_in[11];
    const float* b_hh2  = (const float*)d_in[12];
    const float* W_mos  = (const float*)d_in[13];
    const float* b_mos  = (const float*)d_in[14];

    char* ws = (char*)d_ws;
    size_t o = 0;
    auto carve = [&](size_t bytes) -> char* {
        o = (o + 255) & ~(size_t)255;
        char* p = ws + o;
        o += bytes;
        return p;
    };
    const size_t n_w1  = (size_t)(G1 / 16) * (K1 / 32) * 512;   // 2,359,296
    const size_t n_w2  = (size_t)(G2 / 16) * (K2 / 32) * 512;   //   327,680
    const size_t n_wm  = (size_t)(A_ / 16) * (KM / 32) * 512;   //    16,384
    const size_t n_key = (size_t)N_ * (K_ / 2) * TPAD;          // 16,777,216 dwords
    const size_t n_val = (size_t)N_ * TPAD * V_;                // 33,554,432 bf16
    const size_t n_par = (size_t)(N_ / 16) * NSLICE * PARTW;    // partial records (f32)
    const size_t n_bar = (size_t)(N_ / 16) * L_;                // barrier counters

    __bf16*   w1p   = (__bf16*)carve(n_w1 * 2);
    __bf16*   w2p   = (__bf16*)carve(n_w2 * 2);
    __bf16*   wmp   = (__bf16*)carve(n_wm * 2);
    unsigned* keyT2 = (unsigned*)carve(n_key * 4);
    __bf16*   valT  = (__bf16*)carve(n_val * 2);
    float*    part  = (float*)carve(n_par * 4);
    int*      bar   = (int*)carve(n_bar * 4);

    zero_bar<<<(int)((n_bar + 255) / 256), 256, 0, stream>>>(bar, (int)n_bar);
    pack_wpair<<<(int)((n_w1 + 255) / 256), 256, 0, stream>>>(W_ih1, 640, W_hh1, 512, w1p, G1 / 16, K1 / 32);
    pack_wpair<<<(int)((n_w2 + 255) / 256), 256, 0, stream>>>(W_ih2, 512, W_hh2, 128, w2p, G2 / 16, K2 / 32);
    pack_wpair<<<(int)((n_wm + 255) / 256), 256, 0, stream>>>(W_mos, 256, W_mos, 1,  wmp, A_ / 16, KM / 32);
    tkey<<<(int)((n_key + 255) / 256), 256, 0, stream>>>(key, keyT2);
    tval<<<(int)((n_val + 255) / 256), 256, 0, stream>>>(values, valT);

    decoder_persistent<<<(N_ / 16) * NSLICE, 512, 0, stream>>>(
        lens, text, emb, values, b_ih1, b_hh1, b_ih2, b_hh2, b_mos,
        w1p, w2p, wmp, keyT2, valT, part, bar, (float*)d_out);
}
